// BOWRegressionMulti_89610197664089
// MI455X (gfx1250) — matrix operand order repaired and checked by reading, not compiled, not run on hardware
//
#include <hip/hip_runtime.h>

#define T_LEN   200
#define B_DIM   1024
#define V_DIM   50000
#define L_DIM   512
#define PAD_TOK 1
#define V_PAD   50016                       // V rounded up to multiple of 32 (BK)
#define KSTEPS  (V_PAD / 32)                // 1563 K-steps total
#define SPLITK  4
#define KCHUNK  ((KSTEPS + SPLITK - 1) / SPLITK)   // 391 steps per chunk
#define LDSK    40                          // padded K stride in LDS (16B aligned, bank-spread)

typedef __attribute__((ext_vector_type(16))) __bf16 v16bf;
typedef __attribute__((ext_vector_type(8)))  float  v8f;

union FragBF {
    v16bf v;
    uint4 u[2];
};

// ---------------------------------------------------------------------------
// Kernel 1: zero the bf16 bag-of-words buffer (B_DIM x V_PAD bf16)
// ---------------------------------------------------------------------------
__global__ __launch_bounds__(256) void zero_bow_kernel(unsigned short* __restrict__ bow) {
    const size_t n_chunks = (size_t)B_DIM * V_PAD / 8;   // 8 bf16 per uint4
    uint4* p = reinterpret_cast<uint4*>(bow);
    const uint4 z = {0u, 0u, 0u, 0u};
    for (size_t i = (size_t)blockIdx.x * blockDim.x + threadIdx.x;
         i < n_chunks;
         i += (size_t)gridDim.x * blockDim.x) {
        p[i] = z;
    }
}

// ---------------------------------------------------------------------------
// Kernel 2: scatter bf16 1.0 at bow[b, tok] for every non-PAD token.
// text is [T, B]; duplicate writes store the identical value -> order safe.
// ---------------------------------------------------------------------------
__global__ __launch_bounds__(256) void scatter_bow_kernel(const int* __restrict__ text,
                                                          unsigned short* __restrict__ bow) {
    const int i = blockIdx.x * blockDim.x + threadIdx.x;
    if (i >= T_LEN * B_DIM) return;
    const int b   = i % B_DIM;           // text[t*B + b]
    const int tok = text[i];
    if (tok != PAD_TOK && tok >= 0 && tok < V_DIM) {
        bow[(size_t)b * V_PAD + tok] = 0x3F80;   // bf16 1.0
    }
}

// ---------------------------------------------------------------------------
// Guarded, coalesced load of this thread's share of one W tile (4 x float4).
// Tile = [128 rows (L) x 32 k] f32.  Thread t, j: i4 = t + j*256,
// n = i4>>3 (row), k4 = (i4&7)*4  -> 8 lanes cover one 128B row segment.
// ---------------------------------------------------------------------------
__device__ __forceinline__ void load_w_tile(const float* __restrict__ Wg,
                                            int tileN, int tid, int k0,
                                            float4 (&wf)[4]) {
#pragma unroll
    for (int j = 0; j < 4; ++j) {
        const int i4 = tid + j * 256;
        const int n  = i4 >> 3;
        const int k4 = (i4 & 7) << 2;
        const int gk = k0 + k4;
        const float* p = Wg + (size_t)(tileN * 128 + n) * V_DIM + gk;
        if (gk + 3 < V_DIM) {
            wf[j] = *reinterpret_cast<const float4*>(p);
        } else {
            float4 t = {0.f, 0.f, 0.f, 0.f};
            if (gk + 0 < V_DIM) t.x = p[0];
            if (gk + 1 < V_DIM) t.y = p[1];
            if (gk + 2 < V_DIM) t.z = p[2];
            if (gk + 3 < V_DIM) t.w = p[3];
            wf[j] = t;
        }
    }
}

__device__ __forceinline__ unsigned bfhi(float x) {
    return __float_as_uint(x) >> 16;              // truncated bf16 (hi part)
}
__device__ __forceinline__ unsigned bflo(float x) {
    const float hif = __uint_as_float(__float_as_uint(x) & 0xFFFF0000u);
    return __float_as_uint(x - hif) >> 16;        // truncated bf16 of residual
}

// ---------------------------------------------------------------------------
// Kernel 3: split-K GEMM  partial[c] += bow[1024,V] x W[512,V]^T  via bf16
// WMMA with on-the-fly split-precision W (hi+lo) for ~f32 accuracy.
// Grid: 128 blocks = 8 M-tiles x 4 N-tiles x SPLITK chunks; 256 thr (8 waves).
// Each wave: 32x64 subtile = 2x4 fragments of 16x16, BK=32 per step.
// Register-double-buffered W staging overlaps global loads with WMMAs.
// ---------------------------------------------------------------------------
__global__ __launch_bounds__(256) void bow_gemm_kernel(const unsigned short* __restrict__ bow,
                                                       const float* __restrict__ Wg,
                                                       float* __restrict__ partial) {
    __shared__ unsigned short s_hi[128 * LDSK];
    __shared__ unsigned short s_lo[128 * LDSK];

    const int tid   = threadIdx.x;
    const int wave  = tid >> 5;
    const int lane  = tid & 31;
    const int lhalf = (lane >> 4) & 1;
    const int l15   = lane & 15;

    const int chunk = blockIdx.x & (SPLITK - 1);
    const int tileL = blockIdx.x >> 2;
    const int tileN = tileL & 3;        // 4 tiles over L = 512
    const int tileM = tileL >> 2;       // 8 tiles over B = 1024

    const int wm = wave >> 1;           // 0..3 : M position of wave
    const int wn = wave & 1;            // 0..1 : N position of wave

    const int m_base   = tileM * 128 + wm * 32;
    const int n_base_t = wn * 64;
    const int n_base_g = tileN * 128 + n_base_t;

    const int ks_beg = chunk * KCHUNK;
    const int ks_end = (ks_beg + KCHUNK < KSTEPS) ? (ks_beg + KCHUNK) : KSTEPS;

    v8f acc[2][4] = {};

    // prologue: prefetch first W tile into registers
    float4 wf[4];
    load_w_tile(Wg, tileN, tid, ks_beg * 32, wf);

    for (int ks = ks_beg; ks < ks_end; ++ks) {
        // ---- convert staged registers -> split bf16 hi/lo in LDS ----
#pragma unroll
        for (int j = 0; j < 4; ++j) {
            const int i4 = tid + j * 256;
            const int n  = i4 >> 3;
            const int k4 = (i4 & 7) << 2;
            const float4 w = wf[j];
            uint2 hp, lp;
            hp.x = bfhi(w.x) | (bfhi(w.y) << 16);
            hp.y = bfhi(w.z) | (bfhi(w.w) << 16);
            lp.x = bflo(w.x) | (bflo(w.y) << 16);
            lp.y = bflo(w.z) | (bflo(w.w) << 16);
            *reinterpret_cast<uint2*>(&s_hi[n * LDSK + k4]) = hp;
            *reinterpret_cast<uint2*>(&s_lo[n * LDSK + k4]) = lp;
        }
        __syncthreads();

        const int k0 = ks * 32;

        // ---- A fragments from global bf16 bow (16-bit A layout) ----
        FragBF a[2];
#pragma unroll
        for (int mi = 0; mi < 2; ++mi) {
            const unsigned short* ap =
                bow + (size_t)(m_base + mi * 16 + l15) * V_PAD + k0 + (lhalf ? 16 : 0);
            a[mi].u[0] = *reinterpret_cast<const uint4*>(ap);
            a[mi].u[1] = *reinterpret_cast<const uint4*>(ap + 8);
        }

        // ---- prefetch next K-step's W tile (overlaps with WMMA burst) ----
        if (ks + 1 < ks_end) {
            load_w_tile(Wg, tileN, tid, (ks + 1) * 32, wf);
        }

        // ---- preload ALL B fragments, then the 16-WMMA burst ----
        FragBF bh[4], bl[4];
#pragma unroll
        for (int ni = 0; ni < 4; ++ni) {
            const int n = n_base_t + ni * 16 + l15;
            const unsigned short* ph = &s_hi[n * LDSK + (lhalf ? 16 : 0)];
            const unsigned short* pl = &s_lo[n * LDSK + (lhalf ? 16 : 0)];
            bh[ni].u[0] = *reinterpret_cast<const uint4*>(ph);
            bh[ni].u[1] = *reinterpret_cast<const uint4*>(ph + 8);
            bl[ni].u[0] = *reinterpret_cast<const uint4*>(pl);
            bl[ni].u[1] = *reinterpret_cast<const uint4*>(pl + 8);
        }
#pragma unroll
        for (int ni = 0; ni < 4; ++ni) {
#pragma unroll
            for (int mi = 0; mi < 2; ++mi) {
                acc[mi][ni] = __builtin_amdgcn_wmma_f32_16x16x32_bf16(
                    false, a[mi].v, false, bh[ni].v, (short)0, acc[mi][ni], false, false);
                acc[mi][ni] = __builtin_amdgcn_wmma_f32_16x16x32_bf16(
                    false, a[mi].v, false, bl[ni].v, (short)0, acc[mi][ni], false, false);
            }
        }
        __syncthreads();   // all waves done with LDS before next store phase
    }

    // ---- epilogue: write this chunk's partial tile (disjoint -> no races) ----
    float* pc = partial + (size_t)chunk * B_DIM * L_DIM;
#pragma unroll
    for (int ni = 0; ni < 4; ++ni) {
        const int gn = n_base_g + ni * 16 + l15;
#pragma unroll
        for (int mi = 0; mi < 2; ++mi) {
            const int m0 = m_base + mi * 16 + (lhalf ? 8 : 0);
#pragma unroll
            for (int r = 0; r < 8; ++r) {
                pc[(size_t)(m0 + r) * L_DIM + gn] = acc[mi][ni][r];
            }
        }
    }
}

// ---------------------------------------------------------------------------
// Kernel 4: out = bias + sum_c partial[c]   (deterministic fixed-order sum)
// ---------------------------------------------------------------------------
__global__ __launch_bounds__(256) void reduce_kernel(const float* __restrict__ partial,
                                                     const float* __restrict__ bias,
                                                     float* __restrict__ out) {
    const int f = blockIdx.x * 256 + threadIdx.x;       // index over float4s
    const int col4 = f & (L_DIM / 4 - 1);               // float4 column in [0,128)
    float4 acc = *reinterpret_cast<const float4*>(bias + col4 * 4);
    const size_t stride4 = (size_t)B_DIM * L_DIM / 4;
    const float4* p = reinterpret_cast<const float4*>(partial);
#pragma unroll
    for (int c = 0; c < SPLITK; ++c) {
        const float4 v = p[(size_t)f + c * stride4];
        acc.x += v.x; acc.y += v.y; acc.z += v.z; acc.w += v.w;
    }
    reinterpret_cast<float4*>(out)[f] = acc;
}

// ---------------------------------------------------------------------------
extern "C" void kernel_launch(void* const* d_in, const int* in_sizes, int n_in,
                              void* d_out, int out_size, void* d_ws, size_t ws_size,
                              hipStream_t stream) {
    (void)in_sizes; (void)n_in; (void)out_size; (void)ws_size;

    const int*   text = (const int*)d_in[0];     // [T, B] tokens
    const float* Wg   = (const float*)d_in[1];   // [L, V] f32
    const float* bias = (const float*)d_in[2];   // [L] f32
    float* out        = (float*)d_out;           // [B, L] f32

    // workspace layout:
    //   bow     : B_DIM * V_PAD bf16            (~102.5 MB)
    //   partial : SPLITK * B_DIM * L_DIM f32    (8 MB)
    unsigned short* bow = (unsigned short*)d_ws;
    float* partial = (float*)((char*)d_ws + (size_t)B_DIM * V_PAD * 2);

    zero_bow_kernel<<<2048, 256, 0, stream>>>(bow);
    scatter_bow_kernel<<<(T_LEN * B_DIM + 255) / 256, 256, 0, stream>>>(text, bow);
    bow_gemm_kernel<<<32 * SPLITK, 256, 0, stream>>>(bow, Wg, partial);
    reduce_kernel<<<(B_DIM * L_DIM / 4) / 256, 256, 0, stream>>>(partial, bias, out);
}
